// conGCN_51917564674346
// MI455X (gfx1250) — matrix operand order repaired and checked by reading, not compiled, not run on hardware
//
#include <hip/hip_runtime.h>
#include <math.h>

// ---------------- problem constants ----------------
#define NNODES 20000
#define NFEAT  3000
#define KPAD   3008      // NFEAT padded to multiple of 32 for WMMA K-loop
#define NHID   256
#define NEDGE  640000
#define NOUTC  20
#define BN_EPS 1e-5f

typedef __attribute__((ext_vector_type(16))) __bf16 v16bf;
typedef __attribute__((ext_vector_type(8)))  float  v8f;

__device__ __forceinline__ unsigned short f2bf(float f) {
  unsigned int u = __float_as_uint(f);
  u += 0x7FFFu + ((u >> 16) & 1u);           // round-to-nearest-even
  return (unsigned short)(u >> 16);
}
__device__ __forceinline__ float bf2f(unsigned short h) {
  return __uint_as_float(((unsigned int)h) << 16);
}

// ---------------- conversion kernels ----------------
// activations: [rows,K] f32 -> [rows,KP] bf16 with zero column padding
__global__ void cvt_pad_bf16(const float* __restrict__ src, unsigned short* __restrict__ dst,
                             int rows, int K, int KP) {
  int i = blockIdx.x * 256 + threadIdx.x;
  int total = rows * KP;
  if (i >= total) return;
  int r = i / KP, k = i - r * KP;
  float v = (k < K) ? src[(long long)r * K + k] : 0.f;
  dst[i] = f2bf(v);
}

// weights: [K,256] f32 -> TRANSPOSED [256,KP] bf16 with zero K padding.
// Transposing once here makes every WMMA B-fragment a contiguous 32-byte
// global load (2x global_load_b128), killing the per-K-step LDS stage+barrier.
__global__ void cvt_wT_bf16(const float* __restrict__ src, unsigned short* __restrict__ dst,
                            int K, int KP) {
  int i = blockIdx.x * 256 + threadIdx.x;
  int total = NHID * KP;
  if (i >= total) return;
  int c = i / KP, k = i - c * KP;
  dst[i] = f2bf(k < K ? src[(long long)k * NHID + c] : 0.f);
}

__global__ void zero_f32(float* __restrict__ p, int count) {
  int i = blockIdx.x * 256 + threadIdx.x;
  if (i < count) p[i] = 0.f;
}

// ---------------- bf16 WMMA GEMM (barrier-free) ----------------
// C[M,256] = A[M,Kp](bf16,row-major,lda) @ W[Kp,256] given as BT[256,Kp](bf16,ldb=Kp)
// block: 256 threads = 8 waves; wave owns a 16x64 C slab: 4 independent
// accumulators, 4 back-to-back WMMAs per K-step off one A fragment (4x reuse).
// grid = (Nslabs=4, ceil(Mtiles/8)); x-fastest so column-slabs sharing the same
// A rows run adjacently and re-hit L2 (192 MB) on the A re-read.
__global__ void __launch_bounds__(256)
gemm_bf16_wmma(const unsigned short* __restrict__ A, int lda,
               const unsigned short* __restrict__ BT, int ldb,
               float* __restrict__ C, int ldc,
               int M, int Kp) {
  const int lane   = threadIdx.x & 31;
  const int wave   = threadIdx.x >> 5;
  const int Mtiles = M >> 4;
  const int mtile  = blockIdx.y * 8 + wave;
  if (mtile >= Mtiles) return;                 // wave-uniform: EXEC all-ones below
  const int m0 = mtile << 4;
  const int n0 = blockIdx.x << 6;              // 64-column slab

  // A fragment (contiguous K order, matches B):
  // lanes 0-15: row=lane,    elems0-15 = K+0..15
  // lanes16-31: row=lane-16, elems0-15 = K+16..31
  const int arow = m0 + (lane & 15);
  const int akb  = (lane >> 4) * 16;
  // B fragment: lanes 0-15: col, K+0..15 ; lanes 16-31: col, K+16..31 (elem j = K+j)
  const int bcol = n0 + (lane & 15);
  const int bkb  = (lane >> 4) * 16;

  const unsigned short* ap = A  + (long long)arow * lda + akb;
  const unsigned short* bp = BT + (long long)bcol * ldb + bkb;
  const long long bts = 16ll * ldb;            // BT stride between 16-col tiles

  v8f acc0 = {}, acc1 = {}, acc2 = {}, acc3 = {};
  union Frag { v16bf v; uint4 q[2]; };

  for (int k0 = 0; k0 < Kp; k0 += 32) {
    Frag af, b0, b1, b2, b3;
    af.q[0] = *reinterpret_cast<const uint4*>(ap + k0);
    af.q[1] = *reinterpret_cast<const uint4*>(ap + k0 + 8);
    __builtin_prefetch(ap + k0 + 64, 0, 3);    // global_prefetch_b8, next A slab
    b0.q[0] = *reinterpret_cast<const uint4*>(bp + k0);
    b0.q[1] = *reinterpret_cast<const uint4*>(bp + k0 + 8);
    b1.q[0] = *reinterpret_cast<const uint4*>(bp + bts + k0);
    b1.q[1] = *reinterpret_cast<const uint4*>(bp + bts + k0 + 8);
    b2.q[0] = *reinterpret_cast<const uint4*>(bp + 2 * bts + k0);
    b2.q[1] = *reinterpret_cast<const uint4*>(bp + 2 * bts + k0 + 8);
    b3.q[0] = *reinterpret_cast<const uint4*>(bp + 3 * bts + k0);
    b3.q[1] = *reinterpret_cast<const uint4*>(bp + 3 * bts + k0 + 8);
    acc0 = __builtin_amdgcn_wmma_f32_16x16x32_bf16(false, af.v, false, b0.v,
                                                   (short)0, acc0, false, false);
    acc1 = __builtin_amdgcn_wmma_f32_16x16x32_bf16(false, af.v, false, b1.v,
                                                   (short)0, acc1, false, false);
    acc2 = __builtin_amdgcn_wmma_f32_16x16x32_bf16(false, af.v, false, b2.v,
                                                   (short)0, acc2, false, false);
    acc3 = __builtin_amdgcn_wmma_f32_16x16x32_bf16(false, af.v, false, b3.v,
                                                   (short)0, acc3, false, false);
  }

  // C/D layout: VGPR v, lanes 0-15 -> M=v, lanes 16-31 -> M=8+v; N=lane&15
  const int crow = m0 + (lane >> 4) * 8;
  const int ccol = n0 + (lane & 15);
#pragma unroll
  for (int v = 0; v < 8; ++v) {
    float* cr = C + (long long)(crow + v) * ldc + ccol;
    cr[0]  = acc0[v];
    cr[16] = acc1[v];
    cr[32] = acc2[v];
    cr[48] = acc3[v];
  }
}

// ---------------- SPMM scatter: out[row] += val * dense[col] ----------------
// one thread per (edge, 4-feature chunk); f32 atomics stay in 192MB L2
__global__ void spmm_scatter(const int* __restrict__ rows, const int* __restrict__ cols,
                             const float* __restrict__ vals, int E,
                             const float* __restrict__ dense, float* __restrict__ out) {
  int i = blockIdx.x * 256 + threadIdx.x;
  if (i >= E * 64) return;
  int e = i >> 6;
  int f = (i & 63) << 2;
  int r = rows[e], c = cols[e];
  float v = vals[e];
  float4 d = *reinterpret_cast<const float4*>(dense + (long long)c * NHID + f);
  float* o = out + (long long)r * NHID + f;
  unsafeAtomicAdd(o + 0, v * d.x);
  unsafeAtomicAdd(o + 1, v * d.y);
  unsafeAtomicAdd(o + 2, v * d.z);
  unsafeAtomicAdd(o + 3, v * d.w);
}

// ---------------- BatchNorm stats (per-column sum / sumsq) ----------------
__global__ void bn_stats(const float* __restrict__ src, int Nrows, int rowsPer,
                         float* __restrict__ stats) {
  int col = threadIdx.x;                        // F == 256
  int r0 = blockIdx.x * rowsPer;
  int r1 = min(r0 + rowsPer, Nrows);
  float s = 0.f, s2 = 0.f;
  for (int r = r0; r < r1; ++r) {
    float v = src[(long long)r * NHID + col];
    s += v; s2 += v * v;
  }
  unsafeAtomicAdd(&stats[col], s);
  unsafeAtomicAdd(&stats[NHID + col], s2);
}

// ---------------- BN (training mode) + ELU -> bf16 activation ----------------
// note: pre-BN bias cancels under mean subtraction, so it is omitted entirely
__global__ void bn_elu_bf16(const float* __restrict__ src, const float* __restrict__ stats,
                            const float* __restrict__ gamma, const float* __restrict__ beta,
                            unsigned short* __restrict__ dst, int ldd, int Nrows) {
  int i = blockIdx.x * 256 + threadIdx.x;
  if (i >= Nrows * NHID) return;
  int r = i >> 8, c = i & 255;
  float invN = 1.f / (float)Nrows;
  float m = stats[c] * invN;
  float var = stats[NHID + c] * invN - m * m;
  float y = (src[i] - m) * rsqrtf(var + BN_EPS) * gamma[c] + beta[c];
  y = (y > 0.f) ? y : expm1f(y);                // ELU
  dst[(long long)r * ldd + c] = f2bf(y);
}

// ---------------- final 256->20 GEMM + bias + log_softmax ----------------
__global__ void __launch_bounds__(256)
head_out(const unsigned short* __restrict__ H, const float* __restrict__ W,
         const float* __restrict__ b, float* __restrict__ out, int Nrows) {
  __shared__ float Ws[NHID * NOUTC];
  __shared__ float bs[NOUTC];
  for (int i = threadIdx.x; i < NHID * NOUTC; i += 256) Ws[i] = W[i];
  if (threadIdx.x < NOUTC) bs[threadIdx.x] = b[threadIdx.x];
  __syncthreads();
  int r = blockIdx.x * 256 + threadIdx.x;
  if (r >= Nrows) return;
  float acc[NOUTC];
#pragma unroll
  for (int j = 0; j < NOUTC; ++j) acc[j] = bs[j];
  const unsigned short* h = H + (long long)r * NHID;
  for (int k = 0; k < NHID; ++k) {
    float hv = bf2f(h[k]);
#pragma unroll
    for (int j = 0; j < NOUTC; ++j) acc[j] += hv * Ws[k * NOUTC + j];
  }
  float mx = acc[0];
#pragma unroll
  for (int j = 1; j < NOUTC; ++j) mx = fmaxf(mx, acc[j]);
  float s = 0.f;
#pragma unroll
  for (int j = 0; j < NOUTC; ++j) s += expf(acc[j] - mx);
  float ls = logf(s);
  float* o = out + (long long)r * NOUTC;
#pragma unroll
  for (int j = 0; j < NOUTC; ++j) o[j] = acc[j] - mx - ls;
}

// ---------------- host orchestration ----------------
extern "C" void kernel_launch(void* const* d_in, const int* in_sizes, int n_in,
                              void* d_out, int out_size, void* d_ws, size_t ws_size,
                              hipStream_t stream) {
  (void)in_sizes; (void)n_in; (void)out_size; (void)ws_size;
  const float* x       = (const float*)d_in[0];
  const float* embed   = (const float*)d_in[1];
  const int*   a0i     = (const int*)d_in[2];
  const float* a0v     = (const float*)d_in[3];
  const int*   a1i     = (const int*)d_in[4];
  const float* a1v     = (const float*)d_in[5];
  const int*   a2i     = (const int*)d_in[6];
  const float* a2v     = (const float*)d_in[7];
  const float* Win_exp = (const float*)d_in[8];
  const float* gin_exp = (const float*)d_in[10];
  const float* bein_exp= (const float*)d_in[11];
  const float* Win_sp  = (const float*)d_in[12];
  const float* gin_sp  = (const float*)d_in[14];
  const float* bein_sp = (const float*)d_in[15];
  const float* Win_emb = (const float*)d_in[16];
  const float* gin_emb = (const float*)d_in[18];
  const float* bein_emb= (const float*)d_in[19];
  const float* Wc_exp  = (const float*)d_in[20];
  const float* gc_exp  = (const float*)d_in[22];
  const float* bec_exp = (const float*)d_in[23];
  const float* Wc_sp   = (const float*)d_in[24];
  const float* gc_sp   = (const float*)d_in[26];
  const float* bec_sp  = (const float*)d_in[27];
  const float* Wc_emb  = (const float*)d_in[28];
  const float* gc_emb  = (const float*)d_in[30];
  const float* bec_emb = (const float*)d_in[31];
  const float* Wout1   = (const float*)d_in[32];
  const float* gout1   = (const float*)d_in[34];
  const float* beout1  = (const float*)d_in[35];
  const float* Wf      = (const float*)d_in[36];
  const float* gf      = (const float*)d_in[38];
  const float* bef     = (const float*)d_in[39];
  const float* Wout2   = (const float*)d_in[40];
  const float* bout2   = (const float*)d_in[41];

  // deterministic bump allocator over workspace
  char* wp = (char*)d_ws;
  auto alloc = [&](size_t bytes) -> char* {
    char* r = wp; wp += (bytes + 255) & ~(size_t)255; return r;
  };
  unsigned short* XB      = (unsigned short*)alloc(2ull * NNODES * KPAD);
  unsigned short* EB      = (unsigned short*)alloc(2ull * NNODES * NHID);
  unsigned short* HCAT    = (unsigned short*)alloc(2ull * NNODES * 3 * NHID);
  unsigned short* HB1     = (unsigned short*)alloc(2ull * NNODES * NHID);
  unsigned short* HB2     = (unsigned short*)alloc(2ull * NNODES * NHID);
  float*          MM      = (float*)alloc(4ull * NNODES * NHID);
  float*          SP      = (float*)alloc(4ull * NNODES * NHID);
  float*          STATS   = (float*)alloc(4ull * 2 * NHID);
  // transposed bf16 weights: WT[256, KP]
  unsigned short* WinExpT = (unsigned short*)alloc(2ull * NHID * KPAD);
  unsigned short* WinSpT  = (unsigned short*)alloc(2ull * NHID * KPAD);
  unsigned short* WinEmbT = (unsigned short*)alloc(2ull * NHID * NHID);
  unsigned short* WcExpT  = (unsigned short*)alloc(2ull * 2 * NHID * NHID);
  unsigned short* WcSpT   = (unsigned short*)alloc(2ull * 2 * NHID * NHID);
  unsigned short* WcEmbT  = (unsigned short*)alloc(2ull * 2 * NHID * NHID);
  unsigned short* Wout1T  = (unsigned short*)alloc(2ull * NHID * 3 * NHID);
  unsigned short* WfT     = (unsigned short*)alloc(2ull * NHID * NHID);

  auto blocks = [](int total) { return dim3((unsigned)((total + 255) / 256)); };
  const dim3 T(256);

  auto gemm = [&](const unsigned short* A, int lda, const unsigned short* BT, float* C, int Kp) {
    dim3 g(NHID / 64, (NNODES / 16 + 7) / 8);   // (4, 157), N-slabs fastest
    gemm_bf16_wmma<<<g, T, 0, stream>>>(A, lda, BT, Kp, C, NHID, NNODES, Kp);
  };
  auto spmm = [&](const int* idx, const float* val, const float* dense, float* out) {
    zero_f32<<<blocks(NNODES * NHID), T, 0, stream>>>(out, NNODES * NHID);
    spmm_scatter<<<blocks(NEDGE * 64), T, 0, stream>>>(idx, idx + NEDGE, val, NEDGE, dense, out);
  };
  auto bnelu = [&](const float* src, const float* g, const float* be,
                   unsigned short* dst, int ldd) {
    zero_f32<<<dim3(2), T, 0, stream>>>(STATS, 2 * NHID);
    bn_stats<<<dim3(100), T, 0, stream>>>(src, NNODES, 200, STATS);
    bn_elu_bf16<<<blocks(NNODES * NHID), T, 0, stream>>>(src, STATS, g, be, dst, ldd, NNODES);
  };
  auto cvtwT = [&](const float* src, unsigned short* dst, int K, int KP) {
    cvt_wT_bf16<<<blocks(NHID * KP), T, 0, stream>>>(src, dst, K, KP);
  };

  // ---- per-call conversions to bf16 (weights transposed+padded) ----
  cvt_pad_bf16<<<blocks(NNODES * KPAD), T, 0, stream>>>(x, XB, NNODES, NFEAT, KPAD);
  cvt_pad_bf16<<<blocks(NNODES * NHID), T, 0, stream>>>(embed, EB, NNODES, NHID, NHID);
  cvtwT(Win_exp, WinExpT, NFEAT, KPAD);
  cvtwT(Win_sp,  WinSpT,  NFEAT, KPAD);
  cvtwT(Win_emb, WinEmbT, NHID, NHID);
  for (int i = 0; i < 2; ++i) {
    cvtwT(Wc_exp + i * NHID * NHID, WcExpT + i * NHID * NHID, NHID, NHID);
    cvtwT(Wc_sp  + i * NHID * NHID, WcSpT  + i * NHID * NHID, NHID, NHID);
    cvtwT(Wc_emb + i * NHID * NHID, WcEmbT + i * NHID * NHID, NHID, NHID);
  }
  cvtwT(Wout1, Wout1T, 3 * NHID, 3 * NHID);
  cvtwT(Wf, WfT, NHID, NHID);

  unsigned short* HE = HCAT + 0;          // branch activations live inside the
  unsigned short* HS = HCAT + NHID;       // concat buffer (lda = 768) -> concat is free
  unsigned short* HM = HCAT + 2 * NHID;
  const int LDC3 = 3 * NHID;

  // ---- input GCN layer (exp / sp / embed branches) ----
  gemm(XB, KPAD, WinExpT, MM, KPAD); spmm(a0i, a0v, MM, SP); bnelu(SP, gin_exp, bein_exp, HE, LDC3);
  gemm(XB, KPAD, WinSpT,  MM, KPAD); spmm(a1i, a1v, MM, SP); bnelu(SP, gin_sp,  bein_sp,  HS, LDC3);
  gemm(EB, NHID, WinEmbT, MM, NHID); spmm(a2i, a2v, MM, SP); bnelu(SP, gin_emb, bein_emb, HM, LDC3);

  // ---- L = 2 common hidden GCN layers per branch ----
  for (int i = 0; i < 2; ++i) {
    gemm(HE, LDC3, WcExpT + i * NHID * NHID, MM, NHID);
    spmm(a0i, a0v, MM, SP); bnelu(SP, gc_exp + i * NHID, bec_exp + i * NHID, HE, LDC3);
    gemm(HS, LDC3, WcSpT + i * NHID * NHID, MM, NHID);
    spmm(a1i, a1v, MM, SP); bnelu(SP, gc_sp + i * NHID, bec_sp + i * NHID, HS, LDC3);
    gemm(HM, LDC3, WcEmbT + i * NHID * NHID, MM, NHID);
    spmm(a2i, a2v, MM, SP); bnelu(SP, gc_emb + i * NHID, bec_emb + i * NHID, HM, LDC3);
  }

  // ---- output head: concat(768)->256, FH=1 fc layer, 256->20 + log_softmax ----
  gemm(HCAT, LDC3, Wout1T, MM, LDC3); bnelu(MM, gout1, beout1, HB1, NHID);
  gemm(HB1, NHID, WfT, MM, NHID);     bnelu(MM, gf, bef, HB2, NHID);
  head_out<<<blocks(NNODES), T, 0, stream>>>(HB2, Wout2, bout2, (float*)d_out, NNODES);
}
